// TacotronWithSurprisal_56899726737983
// MI455X (gfx1250) — matrix operand order repaired and checked by reading, not compiled, not run on hardware
//
#include <hip/hip_runtime.h>
#include <hip/hip_bf16.h>
#include <math.h>

#define B_   4096
#define S_   13
#define E_   128
#define HE_  256
#define HD_  256
#define G4   1024            // 4*HE == 4*HD
#define SHD  (S_*HD_)        // 3328
#define F1N  1024
#define NOUT 20000           // 80*250

typedef __bf16 bf16;
typedef __attribute__((ext_vector_type(16))) __bf16 v16bf;
typedef __attribute__((ext_vector_type(8)))  float  v8f;

union FragB16 { v16bf v; uint4 q[2]; };

__device__ __forceinline__ float sigf(float x){ return 1.0f/(1.0f+__expf(-x)); }

// ---------------- weight prep ----------------
__global__ void k_pack_wenc(const float* __restrict__ ew_ih, const float* __restrict__ ew_hh,
                            bf16* __restrict__ out){
  int i = blockIdx.x*256 + threadIdx.x;
  if (i >= 1024*384) return;
  int n = i/384, k = i%384;
  float v = (k < 128) ? ew_ih[n*128 + k] : ew_hh[n*256 + (k-128)];
  out[i] = (bf16)v;
}

__global__ void k_pack_wdec(const float* __restrict__ dw_ih, const float* __restrict__ dw_hh,
                            bf16* __restrict__ out){
  int i = blockIdx.x*256 + threadIdx.x;
  if (i >= 1024*512) return;
  int n = i/512, k = i%512;
  float v = (k < 256) ? dw_ih[n*384 + 128 + k] : dw_hh[n*256 + (k-256)];
  out[i] = (bf16)v;
}

__global__ void k_cvt_bf16(const float* __restrict__ in, bf16* __restrict__ out, int n){
  int i = blockIdx.x*256 + threadIdx.x;
  if (i < n) out[i] = (bf16)in[i];
}

__global__ void k_ebias(const float* __restrict__ a, const float* __restrict__ b,
                        float* __restrict__ out){
  int i = blockIdx.x*256 + threadIdx.x;
  if (i < G4) out[i] = a[i] + b[i];
}

// decoder constant gate vector: db_ih + db_hh + init_in @ dw_ih[:, :128].T
__global__ void k_dvec(const float* __restrict__ db_ih, const float* __restrict__ db_hh,
                       const float* __restrict__ dw_ih, const float* __restrict__ init_in,
                       float* __restrict__ out){
  int n = blockIdx.x*256 + threadIdx.x;
  if (n >= G4) return;
  float acc = db_ih[n] + db_hh[n];
  for (int e = 0; e < E_; ++e) acc += init_in[e] * dw_ih[n*384 + e];
  out[n] = acc;
}

// ---------------- embedding + surprisal projection ----------------
__global__ void k_embed(const int* __restrict__ text, const float* __restrict__ surp,
                        const float* __restrict__ emb, const float* __restrict__ sw,
                        const float* __restrict__ sb, bf16* __restrict__ xb){
  int i = blockIdx.x*256 + threadIdx.x;           // over B*S*E
  if (i >= B_*S_*E_) return;
  int bs = i >> 7, e = i & 127;
  int tok = text[bs];
  float v = emb[tok*E_ + e] + surp[bs]*sw[e] + sb[e];
  xb[i] = (bf16)v;
}

// ---------------- WMMA GEMM: C = [A0|A1] @ W^T + bias ----------------
// A0: (M, K0) bf16 row stride ldA0; A1: (M, K1) bf16 row stride ldA1 (K-concat)
// W:  (N, K0+K1) bf16 row-major (row = output neuron)  -> B[k][n] = W[n][k]
// Block = 128x128 tile, 256 threads (8 waves, 2x4), BK=32.
// Double-buffered LDS fed by GLOBAL_LOAD_ASYNC_TO_LDS_B128 (ASYNCcnt), one
// barrier per K-chunk; async copies for tile i+1 overlap the 8 WMMAs of tile i.
#define BK  32
#define LDT 40   // padded LDS row stride (bf16 elements, keeps 16B alignment)

__global__ __launch_bounds__(256) void k_gemm(
    const bf16* __restrict__ A0, int ldA0, int K0,
    const bf16* __restrict__ A1, int ldA1, int K1,
    const bf16* __restrict__ W,  int N,
    const float* __restrict__ bias,
    float* __restrict__ outF, bf16* __restrict__ outB, int relu)
{
  __shared__ bf16 lsA[2][128*LDT];
  __shared__ bf16 lsB[2][128*LDT];
  const int tid  = threadIdx.x;
  const int lane = tid & 31;
  const int wid  = tid >> 5;
  const int wm   = wid & 1;        // 64-row slab
  const int wn   = wid >> 1;       // 32-col slab
  const int hi   = lane >> 4;
  const int lo   = lane & 15;
  const int m0   = blockIdx.x * 128;
  const int n0   = blockIdx.y * 128;
  const int K    = K0 + K1;
  const int nK   = K / BK;

  v8f zero = {};
  v8f acc[4][2];
  #pragma unroll
  for (int i = 0; i < 4; ++i)
    #pragma unroll
    for (int j = 0; j < 2; ++j) acc[i][j] = zero;

  const int rowL = tid >> 2;          // 0..63
  const int colC = (tid & 3) * 8;     // 0,8,16,24 (bf16 elems; 16B chunks)

  // async-copy one 128x32 A tile + 128x32 W tile into LDS buffer `buf`
  auto issue = [&](int kc, int buf) {
    int k0 = kc * BK;
    #pragma unroll
    for (int p = 0; p < 2; ++p) {
      int r  = rowL + p*64;
      int gm = m0 + r;
      const bf16* asrc = (k0 < K0) ? (A0 + (size_t)gm*ldA0 + k0 + colC)
                                   : (A1 + (size_t)gm*ldA1 + (k0 - K0) + colC);
      unsigned la = (unsigned)(size_t)(&lsA[buf][r*LDT + colC]);   // flat->LDS: addr[31:0]
      asm volatile("global_load_async_to_lds_b128 %0, %1, off"
                   :: "v"(la), "v"((unsigned long long)(size_t)asrc) : "memory");

      int gn = n0 + r;
      int gnc = (gn < N) ? gn : (N - 1);   // clamp: OOB cols never stored
      const bf16* bsrc = W + (size_t)gnc*K + k0 + colC;
      unsigned lb = (unsigned)(size_t)(&lsB[buf][r*LDT + colC]);
      asm volatile("global_load_async_to_lds_b128 %0, %1, off"
                   :: "v"(lb), "v"((unsigned long long)(size_t)bsrc) : "memory");
    }
  };

  issue(0, 0);

  for (int it = 0; it < nK; ++it) {
    // wait for this wave's async copies of tile `it`, then make tile visible
    asm volatile("s_wait_asynccnt 0x0" ::: "memory");
    __syncthreads();
    // prefetch next tile into the other buffer (overlaps with WMMAs below)
    if (it + 1 < nK) issue(it + 1, (it + 1) & 1);

    const bf16* bufA = lsA[it & 1];
    const bf16* bufB = lsB[it & 1];

    FragB16 fa[4], fb[2];
    // A fragment: lane(lo,hi); elems 0..7 -> K 8*hi..8*hi+7, 8..15 -> 16+8*hi..
    #pragma unroll
    for (int mt = 0; mt < 4; ++mt) {
      int r = wm*64 + mt*16 + lo;
      fa[mt].q[0] = *(const uint4*)(&bufA[r*LDT + 8*hi]);
      fa[mt].q[1] = *(const uint4*)(&bufA[r*LDT + 16 + 8*hi]);
    }
    // B fragment: lane n = lo (+half select hi); elems 0..7 -> K 8*hi..8*hi+7, 8..15 -> 16+8*hi..
    #pragma unroll
    for (int nt = 0; nt < 2; ++nt) {
      int c = wn*32 + nt*16 + lo;
      fb[nt].q[0] = *(const uint4*)(&bufB[c*LDT + 8*hi]);
      fb[nt].q[1] = *(const uint4*)(&bufB[c*LDT + 16 + 8*hi]);
    }
    #pragma unroll
    for (int mt = 0; mt < 4; ++mt)
      #pragma unroll
      for (int nt = 0; nt < 2; ++nt)
        acc[mt][nt] = __builtin_amdgcn_wmma_f32_16x16x32_bf16(
            false, fa[mt].v, false, fb[nt].v, (short)0, acc[mt][nt], false, false);
  }

  // epilogue: C/D layout -> lane(lo,hi), vgpr r: row = r + 8*hi, col = lo
  #pragma unroll
  for (int mt = 0; mt < 4; ++mt) {
    #pragma unroll
    for (int nt = 0; nt < 2; ++nt) {
      int gn = n0 + wn*32 + nt*16 + lo;
      if (gn >= N) continue;
      float bv = bias ? bias[gn] : 0.f;
      #pragma unroll
      for (int r = 0; r < 8; ++r) {
        int gm = m0 + wm*64 + mt*16 + hi*8 + r;
        float v = acc[mt][nt][r] + bv;
        if (relu) v = fmaxf(v, 0.f);
        if (outB) outB[(size_t)gm*N + gn] = (bf16)v;
        else      outF[(size_t)gm*N + gn] = v;
      }
    }
  }
}

// ---------------- LSTM cell (shared enc/dec) ----------------
__global__ void k_cell(const float* __restrict__ gates, float* __restrict__ c,
                       float* __restrict__ hf, bf16* __restrict__ hb,
                       float* __restrict__ enc_out_t, bf16* __restrict__ dec_out_t){
  int i = blockIdx.x*256 + threadIdx.x;  // B*256
  if (i >= B_*HE_) return;
  int b = i >> 8, j = i & 255;
  const float* g = gates + (size_t)b*G4;
  float gi = g[j], gf = g[256+j], gg = g[512+j], go = g[768+j];
  float c2 = sigf(gf)*c[i] + sigf(gi)*tanhf(gg);
  float h2 = sigf(go)*tanhf(c2);
  c[i]  = c2;
  hf[i] = h2;
  hb[i] = (bf16)h2;
  if (enc_out_t) enc_out_t[(size_t)b*SHD + j] = h2;
  if (dec_out_t) dec_out_t[(size_t)b*SHD + j] = (bf16)h2;
}

// ---------------- enc_score = enc_out @ aw_e + ab (wave per (b,s)) ----------------
__global__ void k_escore(const float* __restrict__ enc_out, const float* __restrict__ aw,
                         const float* __restrict__ ab, float* __restrict__ esc){
  int gtid = blockIdx.x*256 + threadIdx.x;
  int row  = gtid >> 5;
  int lane = gtid & 31;
  if (row >= B_*S_) return;
  int b = row / S_, s = row % S_;
  const float* e   = enc_out + (size_t)b*SHD + s*HE_;
  const float* awe = aw + HD_;
  float acc = 0.f;
  for (int j = lane; j < HE_; j += 32) acc += e[j]*awe[j];
  #pragma unroll
  for (int off = 16; off > 0; off >>= 1) acc += __shfl_xor(acc, off, 32);
  if (lane == 0) esc[row] = acc + ab[0];
}

// ---------------- attention: softmax over S, context (wave per b) ----------------
__global__ void k_attn(const float* __restrict__ hf, const float* __restrict__ aw,
                       const float* __restrict__ esc, const float* __restrict__ enc_out,
                       bf16* __restrict__ ctx){
  int gtid = blockIdx.x*256 + threadIdx.x;
  int b    = gtid >> 5;
  int lane = gtid & 31;
  if (b >= B_) return;
  const float* h = hf + (size_t)b*HD_;
  float acc = 0.f;
  for (int j = lane; j < HD_; j += 32) acc += h[j]*aw[j];
  #pragma unroll
  for (int off = 16; off > 0; off >>= 1) acc += __shfl_xor(acc, off, 32);

  float sc[S_];
  float m = -1e30f;
  #pragma unroll
  for (int s = 0; s < S_; ++s) { sc[s] = acc + esc[b*S_ + s]; m = fmaxf(m, sc[s]); }
  float sum = 0.f;
  #pragma unroll
  for (int s = 0; s < S_; ++s) { sc[s] = __expf(sc[s] - m); sum += sc[s]; }
  float inv = 1.f/sum;

  const float* eb = enc_out + (size_t)b*SHD;
  for (int j = lane; j < HE_; j += 32) {
    float a = 0.f;
    #pragma unroll
    for (int s = 0; s < S_; ++s) a += sc[s]*eb[s*HE_ + j];
    ctx[(size_t)b*HE_ + j] = (bf16)(a*inv);
  }
}

// ---------------- launch ----------------
extern "C" void kernel_launch(void* const* d_in, const int* in_sizes, int n_in,
                              void* d_out, int out_size, void* d_ws, size_t ws_size,
                              hipStream_t stream) {
  const int*   text  = (const int*)d_in[0];
  const float* surp  = (const float*)d_in[1];
  const float* emb   = (const float*)d_in[2];
  const float* sw    = (const float*)d_in[3];
  const float* sb    = (const float*)d_in[4];
  const float* ew_ih = (const float*)d_in[5];
  const float* ew_hh = (const float*)d_in[6];
  const float* eb_ih = (const float*)d_in[7];
  const float* eb_hh = (const float*)d_in[8];
  const float* aw    = (const float*)d_in[9];
  const float* ab    = (const float*)d_in[10];
  const float* dw_ih = (const float*)d_in[11];
  const float* dw_hh = (const float*)d_in[12];
  const float* db_ih = (const float*)d_in[13];
  const float* db_hh = (const float*)d_in[14];
  const float* f1w   = (const float*)d_in[15];
  const float* f1b   = (const float*)d_in[16];
  const float* f2w   = (const float*)d_in[17];
  const float* f2b   = (const float*)d_in[18];
  const float* initi = (const float*)d_in[19];
  float* out = (float*)d_out;
  (void)in_sizes; (void)n_in; (void)out_size; (void)ws_size;

  char* p = (char*)d_ws;
  auto alloc = [&](size_t bytes) -> char* {
    char* r = p; p += (bytes + 255) & ~(size_t)255; return r;
  };
  bf16*  xb      = (bf16*) alloc((size_t)B_*S_*E_*2);
  bf16*  wenc    = (bf16*) alloc((size_t)1024*384*2);
  bf16*  wdec    = (bf16*) alloc((size_t)1024*512*2);
  bf16*  wf1     = (bf16*) alloc((size_t)F1N*SHD*2);
  bf16*  wf2     = (bf16*) alloc((size_t)NOUT*1024*2);
  float* ebias   = (float*)alloc((size_t)G4*4);
  float* dvec    = (float*)alloc((size_t)G4*4);
  float* gates   = (float*)alloc((size_t)B_*G4*4);
  float* hf      = (float*)alloc((size_t)B_*HE_*4);
  float* cst     = (float*)alloc((size_t)B_*HE_*4);
  bf16*  hb      = (bf16*) alloc((size_t)B_*HE_*2);
  bf16*  ctx     = (bf16*) alloc((size_t)B_*HE_*2);
  float* enc_out = (float*)alloc((size_t)B_*SHD*4);
  float* esc     = (float*)alloc((size_t)B_*S_*4);
  bf16*  dec_out = (bf16*) alloc((size_t)B_*SHD*2);
  bf16*  hfc     = (bf16*) alloc((size_t)B_*F1N*2);

  // --- weight prep (every call; deterministic) ---
  k_pack_wenc<<<(1024*384+255)/256, 256, 0, stream>>>(ew_ih, ew_hh, wenc);
  k_pack_wdec<<<(1024*512+255)/256, 256, 0, stream>>>(dw_ih, dw_hh, wdec);
  k_cvt_bf16<<<(F1N*SHD+255)/256, 256, 0, stream>>>(f1w, wf1, F1N*SHD);
  k_cvt_bf16<<<(NOUT*1024+255)/256, 256, 0, stream>>>(f2w, wf2, NOUT*1024);
  k_ebias<<<(G4+255)/256, 256, 0, stream>>>(eb_ih, eb_hh, ebias);
  k_dvec<<<(G4+255)/256, 256, 0, stream>>>(db_ih, db_hh, dw_ih, initi, dvec);

  // --- input embedding ---
  k_embed<<<(B_*S_*E_+255)/256, 256, 0, stream>>>(text, surp, emb, sw, sb, xb);

  // --- zero recurrent state ---
  hipMemsetAsync(hf,  0, (size_t)B_*HE_*4, stream);
  hipMemsetAsync(cst, 0, (size_t)B_*HE_*4, stream);
  hipMemsetAsync(hb,  0, (size_t)B_*HE_*2, stream);

  dim3 blk(256);
  dim3 gRec(B_/128, G4/128);

  // --- encoder: gates = [x_t | h] @ [ew_ih|ew_hh]^T + (eb_ih+eb_hh) ---
  for (int t = 0; t < S_; ++t) {
    k_gemm<<<gRec, blk, 0, stream>>>(xb + (size_t)t*E_, S_*E_, E_,
                                     hb, HE_, HE_,
                                     wenc, G4, ebias, gates, nullptr, 0);
    k_cell<<<(B_*HE_+255)/256, blk, 0, stream>>>(gates, cst, hf, hb,
                                                 enc_out + (size_t)t*HE_, nullptr);
  }

  // --- attention precompute: enc_score ---
  k_escore<<<(B_*S_*32+255)/256, blk, 0, stream>>>(enc_out, aw, ab, esc);

  // --- decoder: attention -> gates = [ctx | h] @ [dw_ih[:,E:]|dw_hh]^T + dvec ---
  for (int t = 0; t < S_; ++t) {
    k_attn<<<(B_*32+255)/256, blk, 0, stream>>>(hf, aw, esc, enc_out, ctx);
    k_gemm<<<gRec, blk, 0, stream>>>(ctx, HE_, HE_,
                                     hb, HD_, HD_,
                                     wdec, G4, dvec, gates, nullptr, 0);
    k_cell<<<(B_*HE_+255)/256, blk, 0, stream>>>(gates, cst, hf, hb,
                                                 nullptr, dec_out + (size_t)t*HD_);
  }

  // --- FC1: relu(dec_out @ f1w^T + f1b) -> bf16 ---
  k_gemm<<<dim3(B_/128, F1N/128), blk, 0, stream>>>(dec_out, SHD, SHD,
                                                    nullptr, 0, 0,
                                                    wf1, F1N, f1b, nullptr, hfc, 1);

  // --- FC2: hfc @ f2w^T + f2b -> f32 output (B, 80, 250) row-major ---
  k_gemm<<<dim3(B_/128, (NOUT+127)/128), blk, 0, stream>>>(hfc, 1024, 1024,
                                                           nullptr, 0, 0,
                                                           wf2, NOUT, f2b, out, nullptr, 0);
}
